// ConvLSTMCell_82540681494762
// MI455X (gfx1250) — matrix operand order repaired and checked by reading, not compiled, not run on hardware
//
#include <hip/hip_runtime.h>
#include <hip/hip_bf16.h>

typedef __attribute__((ext_vector_type(16))) __bf16 v16bf;
typedef __attribute__((ext_vector_type(8)))  float  v8f;
typedef unsigned int uint32;

#define BB   16
#define HH   64
#define WW   64
#define CIN  64
#define CHID 128
#define HW   (HH*WW)          // 4096
#define OUTSZ (BB*CHID*HW)    // 8388608 elements per output tensor

union Frag { v16bf v; uint32 u[8]; };

// ---------------------------------------------------------------------------
// Pack activation [B,C,H,W] f32 -> bf16 NHWC [B,H,W,C]
// ---------------------------------------------------------------------------
__global__ void pack_nhwc(const float* __restrict__ in, __bf16* __restrict__ out, int C) {
    int idx = blockIdx.x * 256 + threadIdx.x;       // over B*C*H*W
    int chw = C * HW;
    int b  = idx / chw;
    int r  = idx - b * chw;
    int ch = r / HW;
    int yx = r - ch * HW;
    out[(b * HW + yx) * C + ch] = (__bf16)in[idx];
}

// ---------------------------------------------------------------------------
// Pack weight [Co=128, Ci, 3, 3] f32 -> bf16 A row-major [512, 9*Ci],
// K ordered as (tap = ky*3+kx, ci) so each 32-chunk is contiguous in ci.
// ---------------------------------------------------------------------------
__global__ void pack_w(const float* __restrict__ w, __bf16* __restrict__ A, int Ci, int rowOff) {
    int idx = blockIdx.x * 256 + threadIdx.x;       // over 128*Ci*9
    int ci9 = Ci * 9;
    int co  = idx / ci9;
    int r   = idx - co * ci9;
    int ci  = r / 9;
    int tap = r - ci * 9;                            // ky*3+kx
    int Kt  = 9 * Ci;
    A[(rowOff + co) * Kt + tap * Ci + ci] = (__bf16)w[idx];
}

// ---------------------------------------------------------------------------
// Fused implicit-GEMM ConvLSTM cell.
// Block: 16 pixels (one row strip). Wave w owns channels [w*16, w*16+16)
// for ALL FOUR gates (acc tile g = gate g), so the LSTM epilogue is fully
// in-register: no pre-activation LDS buffer needed.
// ---------------------------------------------------------------------------
struct GateParams {
    const float *c, *Wci, *Wcf, *Wco;
    const float *bxi, *bxf, *bxc, *bxo;
    float *out;
};

__global__ __launch_bounds__(256) void convlstm_wmma(
    const __bf16* __restrict__ xp, const __bf16* __restrict__ hp,
    const uint32* __restrict__ Ax, const uint32* __restrict__ Ah,
    GateParams gp)
{
    // One tap's full-Ci im2col slab: 16 pixels x <=128 bf16 (+1 dword skew)
    __shared__ uint32 patch[16][65];

    const int t    = threadIdx.x;
    const int wave = t >> 5;
    const int lane = t & 31;
    const int n    = lane & 15;           // N (pixel) index inside 16x16 tile
    const int half = lane >> 4;           // lane half for K split

    const int p0 = blockIdx.x << 4;       // first pixel of strip
    const int b  = p0 >> 12;              // / (64*64)
    const int y  = (p0 >> 6) & 63;
    const int x0 = p0 & 63;

    const int jj = t >> 4;                // pixel this thread stages (0..15)
    const int q  = t & 15;                // chunk column within pixel row

    v8f acc[4] = {};                      // acc[g] = gate g, 16ch x 16px

    #pragma unroll
    for (int phase = 0; phase < 2; ++phase) {
        const uint32* src = (const uint32*)(phase ? hp : xp);  // NHWC bf16 as dwords
        const uint32* A   = phase ? Ah : Ax;
        const int Ci      = phase ? CHID : CIN;
        const int Kt      = 9 * Ci;
        const int cpt     = Ci >> 5;      // 32-wide K chunks per tap (2 or 4)
        const int dpp     = Ci >> 1;      // dwords per pixel row

        // Hoisted per-gate A row pointers (row = g*128 + wave*16 + n)
        const uint32* Ar[4];
        #pragma unroll
        for (int g = 0; g < 4; ++g)
            Ar[g] = A + (((g << 7) + (wave << 4) + n) * Kt >> 1);

        for (int tap = 0; tap < 9; ++tap) {
            const int dy = tap / 3 - 1;
            const int dx = tap % 3 - 1;
            const int yy = y + dy;
            const int xx = x0 + jj + dx;

            __syncthreads();
            // Stage entire Ci slab for this tap (thread: cpt contiguous dwords)
            if ((unsigned)yy < 64u && (unsigned)xx < 64u) {
                const uint32* rowp = src + (((b << 6) + yy) * 64 + xx) * dpp;
                #pragma unroll
                for (int d = 0; d < 4; ++d)
                    if (d < cpt) patch[jj][q * cpt + d] = rowp[q * cpt + d];
            } else {
                #pragma unroll
                for (int d = 0; d < 4; ++d)
                    if (d < cpt) patch[jj][q * cpt + d] = 0;
            }
            __syncthreads();

            #pragma unroll
            for (int cc = 0; cc < 4; ++cc) {
                if (cc >= cpt) break;
                const int kb = tap * Ci + (cc << 5);       // element K base
                const int kd = (kb >> 1) + (half << 2);    // dword K base

                // B fragment (K x N = 32x16)
                Frag bf;
                #pragma unroll
                for (int v = 0; v < 8; ++v)
                    bf.u[v] = patch[n][(cc << 4) + (half << 2) + v + ((v >> 2) << 2)];

                // 4 gate tiles: A frag = two contiguous 4-dword runs -> b128s
                #pragma unroll
                for (int g = 0; g < 4; ++g) {
                    const uint32* ap = Ar[g] + kd;
                    Frag af;
                    #pragma unroll
                    for (int v = 0; v < 4; ++v) af.u[v] = ap[v];
                    #pragma unroll
                    for (int v = 4; v < 8; ++v) af.u[v] = ap[8 + (v - 4)];
                    acc[g] = __builtin_amdgcn_wmma_f32_16x16x32_bf16(
                        false, af.v, false, bf.v, (short)0, acc[g], false, false);
                }
            }
        }
    }

    // ---------------- In-register LSTM gate epilogue ----------------
    // C/D layout: VGPR v -> channel offset v + half*8, lane -> pixel n.
    const int pix = (y << 6) | (x0 + n);
    #pragma unroll
    for (int v = 0; v < 8; ++v) {
        const int ch   = (wave << 4) + v + (half << 3);
        const int gidx = ((b << 7) + ch) * HW + pix;   // NCHW flat
        const int widx = (ch << 12) + pix;             // [128,64,64] flat

        const float cold = gp.c[gidx];
        const float zi = acc[0][v] + gp.bxi[ch] + gp.Wci[widx] * cold;
        const float zf = acc[1][v] + gp.bxf[ch] + gp.Wcf[widx] * cold;
        const float zc = acc[2][v] + gp.bxc[ch];
        const float zo = acc[3][v] + gp.bxo[ch] + gp.Wco[widx] * cold;

        const float ig = 1.0f / (1.0f + __expf(-zi));
        const float fg = 1.0f / (1.0f + __expf(-zf));
        const float og = 1.0f / (1.0f + __expf(-zo));
        const float cg = tanhf(zc);

        const float cnew = fg * cold + ig * cg;
        const float hnew = og * cnew;

        gp.out[gidx]             = og;     // output
        gp.out[gidx + OUTSZ]     = hnew;   // h_new
        gp.out[gidx + 2 * OUTSZ] = cnew;   // c_new
    }
}

// ---------------------------------------------------------------------------
extern "C" void kernel_launch(void* const* d_in, const int* in_sizes, int n_in,
                              void* d_out, int out_size, void* d_ws, size_t ws_size,
                              hipStream_t stream) {
    const float* x   = (const float*)d_in[0];
    const float* h   = (const float*)d_in[1];
    const float* c   = (const float*)d_in[2];
    const float* Wxi = (const float*)d_in[3];
    const float* bxi = (const float*)d_in[4];
    const float* Whi = (const float*)d_in[5];
    const float* Wci = (const float*)d_in[6];
    const float* Wxf = (const float*)d_in[7];
    const float* bxf = (const float*)d_in[8];
    const float* Whf = (const float*)d_in[9];
    const float* Wcf = (const float*)d_in[10];
    const float* Wxo = (const float*)d_in[11];
    const float* bxo = (const float*)d_in[12];
    const float* Who = (const float*)d_in[13];
    const float* Wco = (const float*)d_in[14];
    const float* Wxc = (const float*)d_in[15];
    const float* bxc = (const float*)d_in[16];
    const float* Whc = (const float*)d_in[17];

    char* ws = (char*)d_ws;
    __bf16* xp = (__bf16*)(ws);                       //  8 MB: x NHWC bf16
    __bf16* hp = (__bf16*)(ws + 8388608);             // 16 MB: h NHWC bf16
    __bf16* Ax = (__bf16*)(ws + 25165824);            // 512 x 576 bf16
    __bf16* Ah = (__bf16*)(ws + 25755648);            // 512 x 1152 bf16

    pack_nhwc<<<(BB*CIN *HW)/256, 256, 0, stream>>>(x, xp, CIN);
    pack_nhwc<<<(BB*CHID*HW)/256, 256, 0, stream>>>(h, hp, CHID);

    pack_w<<<(CHID*CIN *9)/256, 256, 0, stream>>>(Wxi, Ax, CIN,  0);
    pack_w<<<(CHID*CIN *9)/256, 256, 0, stream>>>(Wxf, Ax, CIN,  128);
    pack_w<<<(CHID*CIN *9)/256, 256, 0, stream>>>(Wxc, Ax, CIN,  256);
    pack_w<<<(CHID*CIN *9)/256, 256, 0, stream>>>(Wxo, Ax, CIN,  384);
    pack_w<<<(CHID*CHID*9)/256, 256, 0, stream>>>(Whi, Ah, CHID, 0);
    pack_w<<<(CHID*CHID*9)/256, 256, 0, stream>>>(Whf, Ah, CHID, 128);
    pack_w<<<(CHID*CHID*9)/256, 256, 0, stream>>>(Whc, Ah, CHID, 256);
    pack_w<<<(CHID*CHID*9)/256, 256, 0, stream>>>(Who, Ah, CHID, 384);

    GateParams gp;
    gp.c = c; gp.Wci = Wci; gp.Wcf = Wcf; gp.Wco = Wco;
    gp.bxi = bxi; gp.bxf = bxf; gp.bxc = bxc; gp.bxo = bxo;
    gp.out = (float*)d_out;

    convlstm_wmma<<<(BB*HW)/16, 256, 0, stream>>>(
        xp, hp, (const uint32*)Ax, (const uint32*)Ah, gp);
}
